// GeometricStructureEmbedding_71829033058414
// MI455X (gfx1250) — matrix operand order repaired and checked by reading, not compiled, not run on hardware
//
#include <hip/hip_runtime.h>
#include <hip/hip_fp16.h>

// ---------------------------------------------------------------------------
// GeometricStructureEmbedding for MI455X (gfx1250, wave32, WMMA)
// ---------------------------------------------------------------------------
#define NPTS   320
#define NPAIR  (NPTS * NPTS)        // 102400 (i,j) pairs
#define EMBED  256
#define SIGMA_D_INV 5.0f            // 1 / 0.2
#define FACTOR_A 3.8197186342054885f // 180/(15*pi)
#define GEPS 1e-12f
// dt[t] = 10000^(-t/128) = exp2(-t * log2(10000)/128)
#define NEG_L2_1E4_OVER_128 (-0.1038102529652301f)

typedef __attribute__((ext_vector_type(16))) _Float16 v16h;
typedef __attribute__((ext_vector_type(8)))  float    v8f;

union AV { v16h v; _Float16 h[16]; };

// ---------------------------------------------------------------------------
// K1: 3-NN per point (reference takes top-4 of -dist and drops self, which is
// always nearest) -> store ref vectors refv[i][k][3] = p_nn - p_i
// ---------------------------------------------------------------------------
__global__ void knn_kernel(const float* __restrict__ pts,
                           float* __restrict__ refv) {
    int i = blockIdx.x * blockDim.x + threadIdx.x;
    if (i >= NPTS) return;
    float px = pts[i * 3 + 0], py = pts[i * 3 + 1], pz = pts[i * 3 + 2];
    float b0 = 1e30f, b1 = 1e30f, b2 = 1e30f;
    int   i0 = 0, i1 = 0, i2 = 0;
    for (int j = 0; j < NPTS; ++j) {
        if (j == i) continue;
        float dx = pts[j * 3 + 0] - px;
        float dy = pts[j * 3 + 1] - py;
        float dz = pts[j * 3 + 2] - pz;
        float d = dx * dx + dy * dy + dz * dz;
        if (d < b0)      { b2 = b1; i2 = i1; b1 = b0; i1 = i0; b0 = d; i0 = j; }
        else if (d < b1) { b2 = b1; i2 = i1; b1 = d;  i1 = j; }
        else if (d < b2) { b2 = d;  i2 = j; }
    }
    int nn[3] = { i0, i1, i2 };
    for (int k = 0; k < 3; ++k) {
        int j = nn[k];
        refv[(i * 3 + k) * 3 + 0] = pts[j * 3 + 0] - px;
        refv[(i * 3 + k) * 3 + 1] = pts[j * 3 + 1] - py;
        refv[(i * 3 + k) * 3 + 2] = pts[j * 3 + 2] - pz;
    }
}

// ---------------------------------------------------------------------------
// K2: per-pair scalars: [d_index, angle0, angle1, angle2]
// (order of the 3 angles is irrelevant: a max over that axis follows)
// ---------------------------------------------------------------------------
__global__ void rowscal_kernel(const float* __restrict__ pts,
                               const float* __restrict__ refv,
                               float* __restrict__ rowscal) {
    int r = blockIdx.x * blockDim.x + threadIdx.x;
    if (r >= NPAIR) return;
    int i = r / NPTS, j = r % NPTS;
    float ax = pts[j * 3 + 0] - pts[i * 3 + 0];
    float ay = pts[j * 3 + 1] - pts[i * 3 + 1];
    float az = pts[j * 3 + 2] - pts[i * 3 + 2];
    float sq = ax * ax + ay * ay + az * az;
    float4 o;
    o.x = sqrtf(fmaxf(sq, 0.0f) + GEPS) * SIGMA_D_INV;
    float ang[3];
    #pragma unroll
    for (int k = 0; k < 3; ++k) {
        float rx = refv[(i * 3 + k) * 3 + 0];
        float ry = refv[(i * 3 + k) * 3 + 1];
        float rz = refv[(i * 3 + k) * 3 + 2];
        // cross(r, a)
        float cx = ry * az - rz * ay;
        float cy = rz * ax - rx * az;
        float cz = rx * ay - ry * ax;
        float sv = sqrtf(cx * cx + cy * cy + cz * cz + GEPS);
        float cv = rx * ax + ry * ay + rz * az;
        ang[k] = atan2f(sv, cv) * FACTOR_A;
    }
    o.y = ang[0]; o.z = ang[1]; o.w = ang[2];
    *(float4*)(rowscal + (size_t)r * 4) = o;
}

// ---------------------------------------------------------------------------
// K3: Wt[n][k] = (f16) W[k][n]  -- transposed so B fragments are contiguous
// ---------------------------------------------------------------------------
__global__ void wtrans_kernel(const float* __restrict__ W,
                              _Float16* __restrict__ Wt) {
    int idx = blockIdx.x * blockDim.x + threadIdx.x;
    if (idx >= EMBED * EMBED) return;
    int n = idx / EMBED, k = idx % EMBED;
    Wt[n * EMBED + k] = (_Float16)W[k * EMBED + n];
}

// ---------------------------------------------------------------------------
// K4: fused embedding GEMM.
//   out[r][n] = (sinemb(d_idx[r]) @ Wd)[n] + bd[n]
//             + max_k (sinemb(a_idx[r][k]) @ Wa)[n] + ba[n]
//
// Block = 256 threads = 8 waves; block owns one 16-row tile (rows share the
// same source point i since 320 = 20*16); each wave owns two 16-col tiles.
//
// Phase 1 (cooperative): build all 4 A-matrices (d + 3 angle streams) for the
// 16 rows x 256 K dims in f16 into LDS, already in WMMA A-fragment order:
//   chunk(stream s, kstep st, lane l) = 16 halves, 32B, at
//   Alds[((s*8+st)*32+l)*16]  -- holds row (l&15), K = st*32+(l>=16)*16+{0..15}
// Each of the 256 threads builds one (st, l) chunk for all 4 streams:
// 16 exp2 (stream-invariant) + 64 sin/cos per thread -- 8x less trans work
// than per-wave rebuild, and out of the WMMA loop entirely.
//
// Phase 2: K-loop of 8 steps: 4 ds_load fragments + 4 global B fragments,
// 8 x v_wmma_f32_16x16x32_f16. Epilogue fuses max3 + biases.
// ---------------------------------------------------------------------------
__global__ __launch_bounds__(256) void gse_gemm_kernel(
    const float* __restrict__ rowscal,
    const _Float16* __restrict__ Wtd,
    const _Float16* __restrict__ Wta,
    const float* __restrict__ bd,
    const float* __restrict__ ba,
    float* __restrict__ out) {

    __shared__ _Float16 Alds[4 * 8 * 32 * 16];   // 32 KB

    const int tid  = threadIdx.x;
    const int lane = tid & 31;
    const int wave = tid >> 5;                   // 0..7
    const int m0   = blockIdx.x * 16;            // row-tile base

    // ---------------- Phase 1: cooperative A-panel build -------------------
    {
        const int bstep = tid >> 5;              // k-step this thread builds
        const int blane = tid & 31;              // fragment lane slot
        const int brow  = blane & 15;            // matrix row within tile
        const int bhb   = (blane >> 4) * 16;     // half-wave K offset
        const float4 brs =
            *(const float4*)(rowscal + (size_t)(m0 + brow) * 4);
        const float rsv[4] = { brs.x, brs.y, brs.z, brs.w };

        // t values are stream-invariant: compute the 16 div-terms once
        float dts[16];
        #pragma unroll
        for (int e = 0; e < 16; ++e) {
            const int off = e;
            const int k   = bstep * 32 + bhb + off;
            dts[e] = exp2f((float)(k & 127) * NEG_L2_1E4_OVER_128);
        }

        #pragma unroll
        for (int s = 0; s < 4; ++s) {
            AV f;
            #pragma unroll
            for (int e = 0; e < 16; ++e) {
                const float om = rsv[s] * dts[e];
                // k-step spans one 32-wide band: sin for dims 0..127 (st<4)
                f.h[e] = (_Float16)((bstep < 4) ? __sinf(om) : __cosf(om));
            }
            *(v16h*)&Alds[(((s * 8 + bstep) * 32) + blane) * 16] = f.v;
        }
    }
    __syncthreads();

    // ---------------- Phase 2: WMMA K-loop ---------------------------------
    const int n0 = wave * 32;                    // this wave's 32-col slab
    const int lr = lane & 15;
    const int hi = lane >> 4;
    const int nA = n0 + lr;
    const int nB = n0 + 16 + lr;
    const int kh = hi * 16;                      // B half-wave K offset

    v8f accd0 = {}, accd1 = {};
    v8f acc00 = {}, acc01 = {};
    v8f acc10 = {}, acc11 = {};
    v8f acc20 = {}, acc21 = {};

    #pragma unroll
    for (int st = 0; st < 8; ++st) {
        const int k0 = st * 32;

        const v16h ad = *(const v16h*)&Alds[(((0 * 8 + st) * 32) + lane) * 16];
        const v16h a0 = *(const v16h*)&Alds[(((1 * 8 + st) * 32) + lane) * 16];
        const v16h a1 = *(const v16h*)&Alds[(((2 * 8 + st) * 32) + lane) * 16];
        const v16h a2 = *(const v16h*)&Alds[(((3 * 8 + st) * 32) + lane) * 16];

        const v16h bdA = *(const v16h*)(Wtd + (size_t)nA * EMBED + k0 + kh);
        const v16h bdB = *(const v16h*)(Wtd + (size_t)nB * EMBED + k0 + kh);
        const v16h baA = *(const v16h*)(Wta + (size_t)nA * EMBED + k0 + kh);
        const v16h baB = *(const v16h*)(Wta + (size_t)nB * EMBED + k0 + kh);

        accd0 = __builtin_amdgcn_wmma_f32_16x16x32_f16(false, ad, false, bdA,
                                                       (short)0, accd0, false, false);
        accd1 = __builtin_amdgcn_wmma_f32_16x16x32_f16(false, ad, false, bdB,
                                                       (short)0, accd1, false, false);
        acc00 = __builtin_amdgcn_wmma_f32_16x16x32_f16(false, a0, false, baA,
                                                       (short)0, acc00, false, false);
        acc01 = __builtin_amdgcn_wmma_f32_16x16x32_f16(false, a0, false, baB,
                                                       (short)0, acc01, false, false);
        acc10 = __builtin_amdgcn_wmma_f32_16x16x32_f16(false, a1, false, baA,
                                                       (short)0, acc10, false, false);
        acc11 = __builtin_amdgcn_wmma_f32_16x16x32_f16(false, a1, false, baB,
                                                       (short)0, acc11, false, false);
        acc20 = __builtin_amdgcn_wmma_f32_16x16x32_f16(false, a2, false, baA,
                                                       (short)0, acc20, false, false);
        acc21 = __builtin_amdgcn_wmma_f32_16x16x32_f16(false, a2, false, baB,
                                                       (short)0, acc21, false, false);
    }

    // ---------------- epilogue: out = d + bd + max3(a) + ba ----------------
    const float bdA_ = bd[nA], bdB_ = bd[nB];
    const float baA_ = ba[nA], baB_ = ba[nB];
    #pragma unroll
    for (int v = 0; v < 8; ++v) {
        const int r = m0 + v + hi * 8;           // C/D layout: VGPR v -> row
        const float m3a = fmaxf(fmaxf(acc00[v], acc10[v]), acc20[v]);
        const float m3b = fmaxf(fmaxf(acc01[v], acc11[v]), acc21[v]);
        out[(size_t)r * EMBED + nA] = accd0[v] + bdA_ + m3a + baA_;
        out[(size_t)r * EMBED + nB] = accd1[v] + bdB_ + m3b + baB_;
    }
}

// ---------------------------------------------------------------------------
// Launch
// inputs: points(960) Wd(65536) bd(256) Wa(65536) ba(256); out: 102400*256 f32
// ---------------------------------------------------------------------------
extern "C" void kernel_launch(void* const* d_in, const int* in_sizes, int n_in,
                              void* d_out, int out_size, void* d_ws, size_t ws_size,
                              hipStream_t stream) {
    const float* points = (const float*)d_in[0];
    const float* Wd     = (const float*)d_in[1];
    const float* bd     = (const float*)d_in[2];
    const float* Wa     = (const float*)d_in[3];
    const float* ba     = (const float*)d_in[4];
    float* out = (float*)d_out;

    // workspace layout
    _Float16* Wtd = (_Float16*)d_ws;                  // 256*256 f16 = 128KB
    _Float16* Wta = Wtd + EMBED * EMBED;              // 128KB
    float* refv    = (float*)(Wta + EMBED * EMBED);   // 320*3*3 f32
    float* rowscal = refv + NPTS * 9;                 // 102400*4 f32 (16B aligned)

    knn_kernel<<<(NPTS + 255) / 256, 256, 0, stream>>>(points, refv);
    wtrans_kernel<<<(EMBED * EMBED + 255) / 256, 256, 0, stream>>>(Wd, Wtd);
    wtrans_kernel<<<(EMBED * EMBED + 255) / 256, 256, 0, stream>>>(Wa, Wta);
    rowscal_kernel<<<(NPAIR + 255) / 256, 256, 0, stream>>>(points, refv, rowscal);

    gse_gemm_kernel<<<NPAIR / 16, 256, 0, stream>>>(rowscal, Wtd, Wta, bd, ba, out);
}
